// EncoderProjector_16716012716131
// MI455X (gfx1250) — matrix operand order repaired and checked by reading, not compiled, not run on hardware
//
#include <hip/hip_runtime.h>
#include <hip/hip_bf16.h>

// ---- CDNA5 wave32 WMMA types ----
typedef __bf16 bf16;
typedef bf16  v16bf __attribute__((ext_vector_type(16)));
typedef bf16  v8bf  __attribute__((ext_vector_type(8)));
typedef float v8f   __attribute__((ext_vector_type(8)));

#define T_LEN 4096
#define H     128
#define G3    384
#define DIN   12

static __device__ __forceinline__ bf16 f2bf(float f) {
  return static_cast<bf16>(f);   // gfx1250 native f32->bf16 convert
}

// Native V_TANH_F32 if the toolchain exposes it; else exp+rcp (both single TRANS ops)
#if __has_builtin(__builtin_amdgcn_tanhf)
static __device__ __forceinline__ float tanh_fast(float x) {
  return __builtin_amdgcn_tanhf(x);
}
static __device__ __forceinline__ float sigm(float x) {
  return __builtin_fmaf(0.5f, __builtin_amdgcn_tanhf(0.5f * x), 0.5f);
}
#elif __has_builtin(__builtin_amdgcn_tanh_f32)
static __device__ __forceinline__ float tanh_fast(float x) {
  return __builtin_amdgcn_tanh_f32(x);
}
static __device__ __forceinline__ float sigm(float x) {
  return __builtin_fmaf(0.5f, __builtin_amdgcn_tanh_f32(0.5f * x), 0.5f);
}
#else
static __device__ __forceinline__ float sigm(float x) {
  return __builtin_amdgcn_rcpf(1.0f + __expf(-x));
}
static __device__ __forceinline__ float tanh_fast(float x) {
  return 2.0f * __builtin_amdgcn_rcpf(1.0f + __expf(-2.0f * x)) - 1.0f;
}
#endif

// One workgroup = 16 batch rows of one GRU direction, scanned over all T.
// 8 waves; wave w owns hidden columns [16w, 16w+16) of r, z, n and h.
__global__ __launch_bounds__(256, 1)
void gru_scan_kernel(const float* __restrict__ y,    const float* __restrict__ u,
                     const float* __restrict__ Wi_f, const float* __restrict__ Wh_f,
                     const float* __restrict__ bi_f, const float* __restrict__ bhn_f,
                     const float* __restrict__ Wi_b, const float* __restrict__ Wh_b,
                     const float* __restrict__ bi_b, const float* __restrict__ bhn_b,
                     float* __restrict__ feat)
{
  // DOUBLE-buffered hidden state -> single barrier per timestep
  __shared__ __align__(16) bf16 sh_h[2 * 16 * H];
  __shared__ __align__(16) bf16 sh_x[2 * 16 * 32];  // double-buffered input tile, K padded to 32

  const int g   = blockIdx.x;
  const int dir = g >> 2;            // 0 = forward, 1 = backward
  const int b0  = (g & 3) * 16;      // batch tile origin
  const float* Wi  = dir ? Wi_b  : Wi_f;
  const float* Wh  = dir ? Wh_b  : Wh_f;
  const float* bi  = dir ? bi_b  : bi_f;
  const float* bhn = dir ? bhn_b : bhn_f;

  const int tid    = threadIdx.x;
  const int lane   = tid & 31;
  const int wave   = tid >> 5;        // 0..7
  const int laneHi = (lane >> 4) & 1;
  const int c16    = lane & 15;
  const int hcol   = wave * 16 + c16; // hidden column this lane owns (0..127)

  // ---- zero LDS (h0 = 0, x pad columns 12..31 stay 0 forever) ----
  for (int i = tid; i < 2 * 16 * H;  i += 256) sh_h[i] = f2bf(0.f);
  for (int i = tid; i < 2 * 16 * 32; i += 256) sh_x[i] = f2bf(0.f);

  // ---- preload loop-invariant B fragments (Wh, Wi) into VGPRs as bf16 ----
  // B 32x16 operand as loaded here (contiguous k, wave32): lane holds column N = lane%16;
  // lanes 0-15 hold K = kf*32 + 0..15, lanes 16-31 hold K = kf*32 + 16..31.
  v16bf BR[5], BZ[5], BN[4], BNx;
#pragma unroll
  for (int kf = 0; kf < 4; ++kf) {
#pragma unroll
    for (int f = 0; f < 16; ++f) {
      int k = kf * 32 + laneHi * 16 + f;
      BR[kf][f] = f2bf(Wh[k * G3 +       hcol]);
      BZ[kf][f] = f2bf(Wh[k * G3 + 128 + hcol]);
      BN[kf][f] = f2bf(Wh[k * G3 + 256 + hcol]);
    }
  }
#pragma unroll
  for (int f = 0; f < 16; ++f) {
    int k = laneHi * 16 + f;
    float vr = (k < DIN) ? Wi[k * G3 +       hcol] : 0.f;
    float vz = (k < DIN) ? Wi[k * G3 + 128 + hcol] : 0.f;
    float vn = (k < DIN) ? Wi[k * G3 + 256 + hcol] : 0.f;
    BR[4][f] = f2bf(vr);
    BZ[4][f] = f2bf(vz);
    BNx  [f] = f2bf(vn);
  }
  const float bir  = bi[hcol];
  const float biz  = bi[128 + hcol];
  const float bin  = bi[256 + hcol];
  const float bhnc = bhn[hcol];

  // ---- x-staging role: 192 threads each own one (row, channel) ----
  const int  xr   = tid / DIN;
  const int  xc   = tid - xr * DIN;
  const bool xact = tid < 16 * DIN;
  const int  t0   = dir ? (T_LEN - 1) : 0;

  // per-thread streaming pointer into y or u; advanced by a constant stride.
  // Two-deep register pipeline: xreg holds x[t_{i+1}] during iteration i.
  const float* xptr  = nullptr;
  long         xstep = 0;
  float        xreg  = 0.f;
  if (xact) {
    if (xc < 8) {
      xptr  = y + ((size_t)(b0 + xr) * T_LEN + t0) * 8 + xc;
      xstep = dir ? -8 : 8;
    } else {
      xptr  = u + ((size_t)(b0 + xr) * T_LEN + t0) * 4 + (xc - 8);
      xstep = dir ? -4 : 4;
    }
    sh_x[xr * 32 + xc] = f2bf(xptr[0]);   // stage t0 into buffer 0
    xreg = xptr[xstep];                   // t1 value in register
    xptr += 2 * xstep;                    // now points at t2
  }
  __syncthreads();

  float hold[8] = {0.f, 0.f, 0.f, 0.f, 0.f, 0.f, 0.f, 0.f};
  float hsum[8] = {0.f, 0.f, 0.f, 0.f, 0.f, 0.f, 0.f, 0.f};

  for (int i = 0; i < T_LEN; ++i) {
    bf16* hcur = sh_h + (i & 1) * (16 * H);
    bf16* hnxt = sh_h + ((i + 1) & 1) * (16 * H);
    bf16* xcur = sh_x + (i & 1) * 512;
    bf16* xnxt = sh_x + ((i + 1) & 1) * 512;

    // issue load for x[t_{i+2}] now; it has a whole iteration to complete
    const bool more  = (i + 1) < T_LEN;
    const bool more2 = (i + 2) < T_LEN;
    float xnew = 0.f;
    if (more2 && xact) xnew = *xptr;

    // ---- GEMM: gates = h @ Wh (+ x @ Wi), f32 accumulate ----
    v8f aR = {}; v8f aZ = {}; v8f aN = {}; v8f aNx = {};
#pragma unroll
    for (int kf = 0; kf < 4; ++kf) {
      // A 16x32 bf16, contiguous k order matching B: lane row = lane%16; laneLo K={0..15}, laneHi K={16..31}
      union { v16bf v; v8bf h[2]; } A;
      const v8bf* p = (const v8bf*)(hcur + c16 * H + kf * 32 + laneHi * 16);
      A.h[0] = p[0];
      A.h[1] = p[1];
      aR = __builtin_amdgcn_wmma_f32_16x16x32_bf16(false, A.v, false, BR[kf], (short)0, aR, false, false);
      aZ = __builtin_amdgcn_wmma_f32_16x16x32_bf16(false, A.v, false, BZ[kf], (short)0, aZ, false, false);
      aN = __builtin_amdgcn_wmma_f32_16x16x32_bf16(false, A.v, false, BN[kf], (short)0, aN, false, false);
    }
    {
      union { v16bf v; v8bf h[2]; } A;
      const v8bf* p = (const v8bf*)(xcur + c16 * 32 + laneHi * 16);
      A.h[0] = p[0];
      A.h[1] = p[1];
      aR  = __builtin_amdgcn_wmma_f32_16x16x32_bf16(false, A.v, false, BR[4], (short)0, aR,  false, false);
      aZ  = __builtin_amdgcn_wmma_f32_16x16x32_bf16(false, A.v, false, BZ[4], (short)0, aZ,  false, false);
      aNx = __builtin_amdgcn_wmma_f32_16x16x32_bf16(false, A.v, false, BNx,   (short)0, aNx, false, false);
    }

    // ---- gates: pure register math (r/z/n for this column live in this lane) ----
    bf16 hnew_b[8];
#pragma unroll
    for (int v = 0; v < 8; ++v) {
      float r   = sigm(aR[v] + bir);
      float z   = sigm(aZ[v] + biz);
      float n   = tanh_fast(aNx[v] + bin + r * (aN[v] + bhnc));
      float hn  = n + z * (hold[v] - n);   // (1-z)*n + z*h
      hold[v]   = hn;
      hsum[v]  += hn;
      hnew_b[v] = f2bf(hn);
    }

    // write h_{i+1} into the OTHER buffer -> no pre-store barrier needed.
    // C/D layout: VGPR v, lane -> row = v + 8*laneHi, col = lane%16
#pragma unroll
    for (int v = 0; v < 8; ++v)
      hnxt[(v + laneHi * 8) * H + hcol] = hnew_b[v];
    if (more && xact)
      xnxt[xr * 32 + xc] = f2bf(xreg);
    xreg = xnew;
    if (more2 && xact) xptr += xstep;

    __syncthreads();  // h_{i+1}/x_{i+1} visible; buffers read this iter now reusable
  }

  // ---- time-mean features: feat[b][dir*128 + hcol] ----
  const float inv = 1.0f / (float)T_LEN;
#pragma unroll
  for (int v = 0; v < 8; ++v) {
    int b = b0 + v + laneHi * 8;
    feat[b * 256 + dir * 128 + hcol] = hsum[v] * inv;
  }
}

// Tiny MLP head: one block per batch row. ~13 MFLOP total, irrelevant runtime.
__global__ __launch_bounds__(256, 1)
void mlp_head_kernel(const float* __restrict__ feat,
                     const float* __restrict__ W1, const float* __restrict__ b1,
                     const float* __restrict__ W2, const float* __restrict__ b2,
                     const float* __restrict__ Wout, float* __restrict__ out)
{
  __shared__ float f0[256];
  __shared__ float f1[256];
  __shared__ float f2[128];
  const int b = blockIdx.x, tid = threadIdx.x;

  f0[tid] = feat[b * 256 + tid];
  __syncthreads();

  float a1 = b1[tid];
  for (int k = 0; k < 256; ++k) a1 = fmaf(f0[k], W1[k * 256 + tid], a1);
  f1[tid] = fmaxf(a1, 0.f);
  __syncthreads();

  if (tid < 128) {
    float a2 = b2[tid];
    for (int k = 0; k < 256; ++k) a2 = fmaf(f1[k], W2[k * 128 + tid], a2);
    f2[tid] = fmaxf(a2, 0.f);
  }
  __syncthreads();

  if (tid < 32) {
    float a3 = 0.f;
    for (int k = 0; k < 128; ++k) a3 = fmaf(f2[k], Wout[k * 32 + tid], a3);
    out[b * 32 + tid] = a3;
  }
}

extern "C" void kernel_launch(void* const* d_in, const int* in_sizes, int n_in,
                              void* d_out, int out_size, void* d_ws, size_t ws_size,
                              hipStream_t stream)
{
  const float* y     = (const float*)d_in[0];
  const float* u     = (const float*)d_in[1];
  const float* Wi_f  = (const float*)d_in[2];
  const float* Wh_f  = (const float*)d_in[3];
  const float* bi_f  = (const float*)d_in[4];
  const float* bhn_f = (const float*)d_in[5];
  const float* Wi_b  = (const float*)d_in[6];
  const float* Wh_b  = (const float*)d_in[7];
  const float* bi_b  = (const float*)d_in[8];
  const float* bhn_b = (const float*)d_in[9];
  const float* W1    = (const float*)d_in[10];
  const float* b1    = (const float*)d_in[11];
  const float* W2    = (const float*)d_in[12];
  const float* b2    = (const float*)d_in[13];
  const float* Wout  = (const float*)d_in[14];

  float* feat = (float*)d_ws;       // 64 x 256 f32 time-mean features
  float* out  = (float*)d_out;      // 64 x 32 f32

  gru_scan_kernel<<<dim3(8), dim3(256), 0, stream>>>(
      y, u, Wi_f, Wh_f, bi_f, bhn_f, Wi_b, Wh_b, bi_b, bhn_b, feat);
  mlp_head_kernel<<<dim3(64), dim3(256), 0, stream>>>(
      feat, W1, b1, W2, b2, Wout, out);
}
